// GroupedQueryAttention_88158498718428
// MI455X (gfx1250) — matrix operand order repaired and checked by reading, not compiled, not run on hardware
//
#include <hip/hip_runtime.h>
#include <hip/hip_bf16.h>

typedef __bf16 bf16;
typedef __attribute__((ext_vector_type(16))) __bf16 bf16x16;
typedef __attribute__((ext_vector_type(8)))  float  f32x8;

__device__ inline bf16 f2bf(float f) {
    union { float f; unsigned u; } a; a.f = f;
    unsigned r = a.u + 0x7FFFu + ((a.u >> 16) & 1u);   // round-to-nearest-even
    union { unsigned short s; bf16 b; } o; o.s = (unsigned short)(r >> 16);
    return o.b;
}

__device__ inline bf16x16 mk16(uint4 lo, uint4 hi) {
    union { uint4 u[2]; bf16x16 v; } x; x.u[0] = lo; x.u[1] = hi; return x.v;
}

__device__ inline f32x8 wmma_bf16(bf16x16 a, bf16x16 b, f32x8 c) {
    // (neg_a, A, neg_b, B, c_mod, C, reuse_a, reuse_b)
    return __builtin_amdgcn_wmma_f32_16x16x32_bf16(false, a, false, b, (short)0, c, false, false);
}

// ---------------- elementwise f32 -> bf16 ----------------
__global__ void cvt_bf16_kernel(const float* __restrict__ in, bf16* __restrict__ out, int n) {
    int i = blockIdx.x * 256 + threadIdx.x;
    if (i < n) out[i] = f2bf(in[i]);
}

// ---------------- transpose + convert: in[R][Cc] f32 -> out[Cc][R] bf16 ----------------
__global__ void transpose_cvt_kernel(const float* __restrict__ in, bf16* __restrict__ out,
                                     int R, int Cc) {
    __shared__ float tile[32][33];
    int bx = blockIdx.x;            // along Cc
    int by = blockIdx.y;            // along R
    int x  = bx * 32 + threadIdx.x; // column in input
    int y0 = by * 32 + threadIdx.y; // row base (threadIdx.y in 0..7)
#pragma unroll
    for (int j = 0; j < 32; j += 8)
        tile[threadIdx.y + j][threadIdx.x] = in[(size_t)(y0 + j) * Cc + x];
    __syncthreads();
    int r = by * 32 + threadIdx.x;  // row index in input == column in output
#pragma unroll
    for (int j = 0; j < 32; j += 8)
        out[(size_t)(bx * 32 + threadIdx.y + j) * R + r] = f2bf(tile[threadIdx.x][threadIdx.y + j]);
}

// ---------------- GEMM: C[M][N] f32 = A[M][K] bf16 @ Bt[N][K]^T bf16 ----------------
// block = 128 threads (4 waves); wave computes a 32(M) x 64(N) tile; grid (M/128, N/64)
__global__ __launch_bounds__(128, 1)
void gemm_bf16_nt(const bf16* __restrict__ A, const bf16* __restrict__ Bt,
                  float* __restrict__ C, int M, int N, int K) {
    const int lane = threadIdx.x & 31;
    const int wave = threadIdx.x >> 5;
    const int ln = lane & 15;
    const int lh = lane >> 4;
    const int m0 = blockIdx.x * 128 + wave * 32;
    const int n0 = blockIdx.y * 64;

    f32x8 acc[2][4] = {};
    for (int kb = 0; kb < K; kb += 32) {
        bf16x16 a[2], b[4];
#pragma unroll
        for (int mt = 0; mt < 2; ++mt) {
            const bf16* ar = A + (size_t)(m0 + mt * 16 + ln) * K + kb;
            a[mt] = mk16(*reinterpret_cast<const uint4*>(ar + lh * 16),
                         *reinterpret_cast<const uint4*>(ar + lh * 16 + 8));
        }
#pragma unroll
        for (int nt = 0; nt < 4; ++nt) {
            const bf16* br = Bt + (size_t)(n0 + nt * 16 + ln) * K + kb + lh * 16;
            b[nt] = mk16(*reinterpret_cast<const uint4*>(br),
                         *reinterpret_cast<const uint4*>(br + 8));
        }
#pragma unroll
        for (int mt = 0; mt < 2; ++mt)
#pragma unroll
            for (int nt = 0; nt < 4; ++nt)
                acc[mt][nt] = wmma_bf16(a[mt], b[nt], acc[mt][nt]);
    }
#pragma unroll
    for (int mt = 0; mt < 2; ++mt)
#pragma unroll
        for (int nt = 0; nt < 4; ++nt)
#pragma unroll
            for (int r = 0; r < 8; ++r)
                C[(size_t)(m0 + mt * 16 + r + 8 * lh) * N + n0 + nt * 16 + ln] = acc[mt][nt][r];
}

// ---------------- per-head LayerNorm (width 128, weight only) f32 -> bf16 ----------------
// one wave per (row, head); block = 256 threads = 8 waves
__global__ void ln_head_kernel(const float* __restrict__ X, const float* __restrict__ gamma,
                               bf16* __restrict__ Y, int rows, int H) {
    int w = blockIdx.x * 8 + (threadIdx.x >> 5);
    int lane = threadIdx.x & 31;
    if (w >= rows) return;
    int Cw = H * 128;
    int t = w / H, h = w % H;
    const float* xp = X + (size_t)t * Cw + h * 128 + lane * 4;
    float4 x = *reinterpret_cast<const float4*>(xp);
    float s = x.x + x.y + x.z + x.w;
#pragma unroll
    for (int mk = 16; mk >= 1; mk >>= 1) s += __shfl_xor(s, mk);
    float mu = s * (1.0f / 128.0f);
    float d0 = x.x - mu, d1 = x.y - mu, d2 = x.z - mu, d3 = x.w - mu;
    float ss = d0 * d0 + d1 * d1 + d2 * d2 + d3 * d3;
#pragma unroll
    for (int mk = 16; mk >= 1; mk >>= 1) ss += __shfl_xor(ss, mk);
    float rs = rsqrtf(ss * (1.0f / 128.0f) + 1e-5f);
    float4 g = *reinterpret_cast<const float4*>(gamma + lane * 4);
    union { bf16 b[4]; uint2 u; } o;
    o.b[0] = f2bf(d0 * rs * g.x);
    o.b[1] = f2bf(d1 * rs * g.y);
    o.b[2] = f2bf(d2 * rs * g.z);
    o.b[3] = f2bf(d3 * rs * g.w);
    *reinterpret_cast<uint2*>(Y + (size_t)t * Cw + h * 128 + lane * 4) = o.u;
}

// ---------------- flash-style causal GQA attention ----------------
// q: [T][2048] bf16 (LN'd), kmat: [T][512] bf16 (LN'd), vt: [4][128][T] bf16
// O: [T][2048] f32. block = 128 threads (4 waves); wave owns 16 query rows.
// grid = (T/64, 16 heads)
__global__ __launch_bounds__(128, 1)
void attn_kernel(const bf16* __restrict__ q, const bf16* __restrict__ kmat,
                 const bf16* __restrict__ vt, const float* __restrict__ temp,
                 float* __restrict__ O) {
    const int lane = threadIdx.x & 31;
    const int wave = threadIdx.x >> 5;
    const int ln = lane & 15;
    const int lh = lane >> 4;
    const int t0 = (blockIdx.x * 4 + wave) * 16;
    const int h = blockIdx.y;
    const int kvh = h >> 2;
    const float scale = fminf(__expf(temp[0]), 50.0f);

    __shared__ alignas(16) bf16 pbuf_all[4][16 * 40];   // row stride 40 -> conflict-free b128 reads
    bf16* pbuf = pbuf_all[wave];

    // Q fragments: 16 rows x 128 d  =  4 fragments of 16x32
    bf16x16 qf[4];
    {
        const bf16* qrow = q + (size_t)(t0 + ln) * 2048 + h * 128;
#pragma unroll
        for (int kb = 0; kb < 4; ++kb) {
            qf[kb] = mk16(*reinterpret_cast<const uint4*>(qrow + kb * 32 + lh * 16),
                          *reinterpret_cast<const uint4*>(qrow + kb * 32 + lh * 16 + 8));
        }
    }

    f32x8 oa[8] = {};
    float mrow[8], lrow[8];
#pragma unroll
    for (int r = 0; r < 8; ++r) { mrow[r] = -1e30f; lrow[r] = 0.0f; }

    const bf16* vhead = vt + (size_t)kvh * 128 * 2048;

    for (int s0 = 0; s0 < t0 + 16; s0 += 32) {
        // scores: two adjacent 16x16 tiles over d = 128
        f32x8 sc0 = {}, sc1 = {};
#pragma unroll
        for (int kb = 0; kb < 4; ++kb) {
            const bf16* kr0 = kmat + (size_t)(s0 + ln) * 512 + kvh * 128 + kb * 32 + lh * 16;
            const bf16* kr1 = kmat + (size_t)(s0 + 16 + ln) * 512 + kvh * 128 + kb * 32 + lh * 16;
            bf16x16 b0 = mk16(*reinterpret_cast<const uint4*>(kr0),
                              *reinterpret_cast<const uint4*>(kr0 + 8));
            bf16x16 b1 = mk16(*reinterpret_cast<const uint4*>(kr1),
                              *reinterpret_cast<const uint4*>(kr1 + 8));
            sc0 = wmma_bf16(qf[kb], b0, sc0);
            sc1 = wmma_bf16(qf[kb], b1, sc1);
        }

        // online softmax: row r of the C tile lives in vector slot r, lanes of this half
        float af[8];
#pragma unroll
        for (int r = 0; r < 8; ++r) {
            const int trow = t0 + r + 8 * lh;
            float x0 = sc0[r] * scale;
            float x1 = sc1[r] * scale;
            if (s0 + ln > trow)      x0 = -1e30f;
            if (s0 + 16 + ln > trow) x1 = -1e30f;
            float mx = fmaxf(x0, x1);
#pragma unroll
            for (int mk = 8; mk >= 1; mk >>= 1) mx = fmaxf(mx, __shfl_xor(mx, mk));
            float mnew = fmaxf(mrow[r], mx);
            float alpha = __expf(mrow[r] - mnew);
            float p0 = __expf(x0 - mnew);
            float p1 = __expf(x1 - mnew);
            float rsum = p0 + p1;
#pragma unroll
            for (int mk = 8; mk >= 1; mk >>= 1) rsum += __shfl_xor(rsum, mk);
            lrow[r] = lrow[r] * alpha + rsum;
            mrow[r] = mnew;
            af[r] = alpha;
            pbuf[(r + 8 * lh) * 40 + ln] = f2bf(p0);
            pbuf[(r + 8 * lh) * 40 + 16 + ln] = f2bf(p1);
        }
#pragma unroll
        for (int dt = 0; dt < 8; ++dt)
#pragma unroll
            for (int r = 0; r < 8; ++r) oa[dt][r] *= af[r];

        asm volatile("s_wait_dscnt 0x0" ::: "memory");

        // re-read P (16x32) in A-fragment layout from LDS
        const bf16* prow = pbuf + ln * 40 + lh * 16;
        bf16x16 pf = mk16(*reinterpret_cast<const uint4*>(prow),
                          *reinterpret_cast<const uint4*>(prow + 8));

#pragma unroll
        for (int dt = 0; dt < 8; ++dt) {
            const bf16* vr = vhead + (size_t)(dt * 16 + ln) * 2048 + s0 + lh * 16;
            bf16x16 bv = mk16(*reinterpret_cast<const uint4*>(vr),
                              *reinterpret_cast<const uint4*>(vr + 8));
            oa[dt] = wmma_bf16(pf, bv, oa[dt]);
        }
    }

    float inv[8];
#pragma unroll
    for (int r = 0; r < 8; ++r) inv[r] = 1.0f / lrow[r];
#pragma unroll
    for (int dt = 0; dt < 8; ++dt)
#pragma unroll
        for (int r = 0; r < 8; ++r)
            O[(size_t)(t0 + r + 8 * lh) * 2048 + h * 128 + dt * 16 + ln] = oa[dt][r] * inv[r];
}

extern "C" void kernel_launch(void* const* d_in, const int* in_sizes, int n_in,
                              void* d_out, int out_size, void* d_ws, size_t ws_size,
                              hipStream_t stream) {
    (void)in_sizes; (void)n_in; (void)out_size; (void)ws_size;
    const int T = 2048, Cd = 2048, KV = 512, H = 16, HK = 4;

    const float* x    = (const float*)d_in[0];
    const float* Wq   = (const float*)d_in[1];
    const float* Wk   = (const float*)d_in[2];
    const float* Wv   = (const float*)d_in[3];
    const float* Wo   = (const float*)d_in[4];
    const float* gq   = (const float*)d_in[5];
    const float* gk   = (const float*)d_in[6];
    const float* temp = (const float*)d_in[7];

    char* w = (char*)d_ws;
    const size_t MB = 1ull << 20;
    bf16*  xb   = (bf16*)(w + 0 * MB);     //  8 MB  x bf16
    bf16*  Wqt  = (bf16*)(w + 8 * MB);     //  8 MB  Wq^T bf16
    bf16*  Wkt  = (bf16*)(w + 16 * MB);    //  2 MB
    bf16*  Wvt  = (bf16*)(w + 18 * MB);    //  2 MB
    bf16*  Wot  = (bf16*)(w + 20 * MB);    //  8 MB
    float* Qf   = (float*)(w + 28 * MB);   // 16 MB  Q f32 (reused as O f32)
    float* Kf   = (float*)(w + 44 * MB);   //  4 MB
    float* Vf   = (float*)(w + 48 * MB);   //  4 MB
    bf16*  qb   = (bf16*)(w + 52 * MB);    //  8 MB  LN'd Q bf16
    bf16*  kb_  = (bf16*)(w + 60 * MB);    //  2 MB  LN'd K bf16
    bf16*  vtb  = (bf16*)(w + 62 * MB);    //  2 MB  V^T bf16
    float* Of   = Qf;                      // alias: Q f32 consumed by LN before attention
    bf16*  ob   = (bf16*)(w + 44 * MB);    // alias: K/V f32 consumed before O convert

    dim3 tb32x8(32, 8);

    // 1) convert x, transpose+convert weights
    cvt_bf16_kernel<<<(T * Cd) / 256, 256, 0, stream>>>(x, xb, T * Cd);
    transpose_cvt_kernel<<<dim3(Cd / 32, Cd / 32), tb32x8, 0, stream>>>(Wq, Wqt, Cd, Cd);
    transpose_cvt_kernel<<<dim3(KV / 32, Cd / 32), tb32x8, 0, stream>>>(Wk, Wkt, Cd, KV);
    transpose_cvt_kernel<<<dim3(KV / 32, Cd / 32), tb32x8, 0, stream>>>(Wv, Wvt, Cd, KV);
    transpose_cvt_kernel<<<dim3(Cd / 32, Cd / 32), tb32x8, 0, stream>>>(Wo, Wot, Cd, Cd);

    // 2) projections (WMMA bf16, f32 accumulate)
    gemm_bf16_nt<<<dim3(T / 128, Cd / 64), 128, 0, stream>>>(xb, Wqt, Qf, T, Cd, Cd);
    gemm_bf16_nt<<<dim3(T / 128, KV / 64), 128, 0, stream>>>(xb, Wkt, Kf, T, KV, Cd);
    gemm_bf16_nt<<<dim3(T / 128, KV / 64), 128, 0, stream>>>(xb, Wvt, Vf, T, KV, Cd);

    // 3) per-head LayerNorm -> bf16; transpose V -> vt[d][s] bf16
    ln_head_kernel<<<(T * H) / 8, 256, 0, stream>>>(Qf, gq, qb, T * H, H);
    ln_head_kernel<<<(T * HK) / 8, 256, 0, stream>>>(Kf, gk, kb_, T * HK, HK);
    transpose_cvt_kernel<<<dim3(KV / 32, T / 32), tb32x8, 0, stream>>>(Vf, vtb, T, KV);

    // 4) causal GQA flash attention (WMMA for q.k^T and P.v)
    attn_kernel<<<dim3(T / 64, H), 128, 0, stream>>>(qb, kb_, vtb, temp, Of);

    // 5) output projection
    cvt_bf16_kernel<<<(T * Cd) / 256, 256, 0, stream>>>(Of, ob, T * Cd);
    gemm_bf16_nt<<<dim3(T / 128, Cd / 64), 128, 0, stream>>>(ob, Wot, (float*)d_out, T, Cd, Cd);
}
